// ResGraphModule_79121887527181
// MI455X (gfx1250) — matrix operand order repaired and checked by reading, not compiled, not run on hardware
//
#include <hip/hip_runtime.h>

// ---------------------------------------------------------------------------
// SchNet-style residual graph block on MI455X (gfx1250), bf16 WMMA pipeline.
// All GEMMs run on v_wmma_f32_16x16x32_bf16 (fp32 accumulate). Dimensions
// (C=128, NF=256, EC=64, EC+C=192; N,E multiples of 16) tile exactly.
// Scatter (segment_sum) via global_atomic_add_f32 into an L2-resident
// agg[N,256] buffer (51 MB << 192 MB L2). The h[col] gather uses gfx1250
// async global->LDS loads (ASYNCcnt) when the builtin is available.
// ---------------------------------------------------------------------------

#define C_CH   128   // in/out channels
#define NF_CH  256   // SchNet num_filters
#define EC_CH  64    // edge channels (RBF)
#define EIN_CH 192   // EC + C

typedef __attribute__((ext_vector_type(16))) __bf16 v16bf;
typedef __attribute__((ext_vector_type(4)))  __bf16 v4bf;
typedef __attribute__((ext_vector_type(8)))  float  v8f;
typedef int v4i_vs __attribute__((vector_size(4 * sizeof(int))));

#if __has_builtin(__builtin_amdgcn_global_load_async_to_lds_b128) && \
    __has_builtin(__builtin_amdgcn_s_wait_asynccnt)
#define HAVE_ASYNC_LDS 1
#else
#define HAVE_ASYNC_LDS 0
#endif

#if HAVE_ASYNC_LDS
// 16-byte async global->LDS copy; imm offset applies to BOTH addresses
// (ISA 15.18). Offset must be an immediate -> template parameter.
template <int OFF>
__device__ __forceinline__ void async_copy16(const void* gsrc, void* ldst) {
    __builtin_amdgcn_global_load_async_to_lds_b128(
        (__attribute__((address_space(1))) v4i_vs*)gsrc,
        (__attribute__((address_space(3))) v4i_vs*)ldst, OFF, 0);
}
#endif

// shifted softplus: softplus(x) - ln(2)
__device__ __forceinline__ float sspf(float v) {
    float s = (v > 15.0f) ? v : log1pf(expf(v));
    return s - 0.6931471805599453f;
}

// A fragment: 16x32 bf16 tile, row-major with leading dim `ld` (elements).
// lane(0-15): row M=lane, K chunk base 0;  lane(16-31): row M=lane-16, K base 8.
// elems 0..7 -> K = kb..kb+7 ; elems 8..15 -> K = kb+16..kb+23  (ISA 7.12.2)
__device__ __forceinline__ v16bf load_a(const __bf16* base, int ld) {
    int lane = threadIdx.x & 31;
    int m  = lane & 15;
    int kb = (lane >> 4) << 3;
    const __bf16* p = base + m * ld + kb;
    union { v16bf v; uint4 q[2]; } u;
    u.q[0] = *(const uint4*)(p);
    u.q[1] = *(const uint4*)(p + 16);
    return u.v;
}
// B fragment: 32x16 bf16 tile of a K-major (row-major KxN) matrix with
// leading dim `ld`. lane: column N=lane%16; elems follow A's K order.
__device__ __forceinline__ v16bf load_b(const __bf16* base, int ld) {
    int lane = threadIdx.x & 31;
    int n  = lane & 15;
    int kb = (lane >> 4) << 3;
    v16bf v;
#pragma unroll
    for (int s = 0; s < 16; ++s) {
        int k = (s < 8) ? (kb + s) : (16 + kb + s - 8);
        v[s] = base[k * ld + n];
    }
    return v;
}
__device__ __forceinline__ v8f wmma_bf(v16bf a, v16bf b, v8f c) {
    return __builtin_amdgcn_wmma_f32_16x16x32_bf16(
        /*neg_a=*/false, a, /*neg_b=*/false, b,
        /*c_mod=*/(short)0, c, /*reuse_a=*/false, /*reuse_b=*/false);
}

// --------------------------- prep kernels ----------------------------------
// 4-wide f32 -> bf16 convert (native v_cvt; all sizes are multiples of 4)
__global__ void k_f32_to_bf16(const float* __restrict__ src,
                              __bf16* __restrict__ dst, int n) {
    int i = (blockIdx.x * 256 + threadIdx.x) * 4;
    if (i >= n) return;
    float4 f = *(const float4*)(src + i);
    union { v4bf v; unsigned long long u; } u;
    u.v[0] = (__bf16)f.x; u.v[1] = (__bf16)f.y;
    u.v[2] = (__bf16)f.z; u.v[3] = (__bf16)f.w;
    *(unsigned long long*)(dst + i) = u.u;
}
__global__ void k_zero_f32(float* __restrict__ p, int n) {
    int i = blockIdx.x * 256 + threadIdx.x;
    if (i < n) p[i] = 0.0f;
}

// --------------------- h = x @ W_lin1  -> bf16 [N, NF] ---------------------
// block = 128 threads (4 waves); each block: one 16-row tile; wave handles
// 4 of the 16 N-tiles; K loop = 128/32 = 4 WMMAs per tile.
__global__ void __launch_bounds__(128)
k_lin1(const __bf16* __restrict__ x_bf,
       const __bf16* __restrict__ w1,          // [128][256] bf16 K-major
       __bf16* __restrict__ h_bf) {
    int rt   = blockIdx.x;
    int wave = threadIdx.x >> 5;
    int lane = threadIdx.x & 31;
    const __bf16* abase = x_bf + (size_t)rt * 16 * C_CH;
    int n16 = lane & 15, mh = (lane >> 4) * 8;
    for (int nt = wave; nt < NF_CH / 16; nt += 4) {
        v8f acc = {};
#pragma unroll
        for (int kt = 0; kt < C_CH / 32; ++kt) {
            v16bf a = load_a(abase + kt * 32, C_CH);
            v16bf b = load_b(w1 + (size_t)kt * 32 * NF_CH + nt * 16, NF_CH);
            acc = wmma_bf(a, b, acc);
        }
        int n = nt * 16 + n16;
#pragma unroll
        for (int v = 0; v < 8; ++v)
            h_bf[((size_t)(rt * 16 + v + mh)) * NF_CH + n] = (__bf16)acc[v];
    }
}

// ---- fused edge filter net + modulate + scatter-sum (the hot kernel) ------
// block = 256 threads (8 waves), one 16-edge tile per block.
// stage1: t1 = ssp(edge_attr @ W_f1 + b_f1)        [16,256]  (LDS, bf16)
// stage2: Wf = t1 @ W_f2 + b_f2 ; msg = h[col]*Wf ; atomicAdd agg[row]
__global__ void __launch_bounds__(256)
k_edge_cfconv(const float* __restrict__ edge_attr,
              const int*   __restrict__ eidx,          // [2,E] flat
              const __bf16* __restrict__ h_bf,
              const __bf16* __restrict__ wf1, const float* __restrict__ bf1,
              const __bf16* __restrict__ wf2, const float* __restrict__ bf2,
              float* __restrict__ agg, int E) {
    __shared__ __align__(16) __bf16 s_ea[16 * EC_CH];  // 2 KB
    __shared__ __align__(16) __bf16 s_t1[16 * NF_CH];  // 8 KB
    __shared__ __align__(16) __bf16 s_h [16 * NF_CH];  // 8 KB
    __shared__ int s_row[16];

    int e0 = blockIdx.x * 16;
    int t  = threadIdx.x;

    // gather h[col[m]] rows into LDS (32 B per thread)
    {
        int m = t >> 4, ch = t & 15;
        int cm = eidx[E + e0 + m];
#if HAVE_ASYNC_LDS
        const __bf16* src = h_bf + (size_t)cm * NF_CH + ch * 16;
        __bf16* dst = s_h + m * NF_CH + ch * 16;
        async_copy16<0>(src, dst);
        async_copy16<16>(src, dst);
#else
        const uint4* src = (const uint4*)(h_bf + (size_t)cm * NF_CH);
        uint4* dst = (uint4*)(s_h + m * NF_CH);
        dst[ch * 2]     = src[ch * 2];
        dst[ch * 2 + 1] = src[ch * 2 + 1];
#endif
    }
    // warm the L2/WGP$ path for the hot 128 KB W_f2 (global_prefetch_b8)
    __builtin_prefetch(wf2 + t * 256, 0, 1);
    // edge_attr tile -> bf16 LDS
    for (int i = t; i < 16 * EC_CH; i += 256)
        s_ea[i] = (__bf16)edge_attr[(size_t)e0 * EC_CH + i];
    if (t < 16) s_row[t] = eidx[e0 + t];
#if HAVE_ASYNC_LDS
    __builtin_amdgcn_s_wait_asynccnt(0);
#endif
    __syncthreads();

    int wave = t >> 5, lane = t & 31;
    int n16 = lane & 15, mh = (lane >> 4) * 8;

    // stage 1
#pragma unroll
    for (int i = 0; i < 2; ++i) {
        int nt = wave + i * 8;
        v8f acc = {};
#pragma unroll
        for (int kt = 0; kt < EC_CH / 32; ++kt) {
            v16bf a = load_a(s_ea + kt * 32, EC_CH);
            v16bf b = load_b(wf1 + (size_t)kt * 32 * NF_CH + nt * 16, NF_CH);
            acc = wmma_bf(a, b, acc);
        }
        int n = nt * 16 + n16;
        float bias = bf1[n];
#pragma unroll
        for (int v = 0; v < 8; ++v)
            s_t1[(v + mh) * NF_CH + n] = (__bf16)sspf(acc[v] + bias);
    }
    __syncthreads();

    // stage 2 + modulate + scatter
#pragma unroll
    for (int i = 0; i < 2; ++i) {
        int nt = wave + i * 8;
        v8f acc = {};
#pragma unroll
        for (int kt = 0; kt < NF_CH / 32; ++kt) {
            v16bf a = load_a(s_t1 + kt * 32, NF_CH);
            v16bf b = load_b(wf2 + (size_t)kt * 32 * NF_CH + nt * 16, NF_CH);
            acc = wmma_bf(a, b, acc);
        }
        int n = nt * 16 + n16;
        float bias = bf2[n];
#pragma unroll
        for (int v = 0; v < 8; ++v) {
            int m = v + mh;
            float msg = (acc[v] + bias) * (float)s_h[m * NF_CH + n];
            atomicAdd(&agg[(size_t)s_row[m] * NF_CH + n], msg);
        }
    }
}

// ---- x_out = relu(ssp(agg@W_lin2 + b)@W_lin3 + b) + x ---------------------
// block = 256 (8 waves), one 16-node tile; wave handles one N-tile per stage.
__global__ void __launch_bounds__(256)
k_node_out(const float* __restrict__ agg,
           const __bf16* __restrict__ w2, const float* __restrict__ b2,
           const __bf16* __restrict__ w3, const float* __restrict__ b3,
           const float* __restrict__ x, float* __restrict__ xout) {
    __shared__ __align__(16) __bf16 s_a[16 * NF_CH];  // 8 KB
    __shared__ __align__(16) __bf16 s_t[16 * C_CH];   // 4 KB
    int r0 = blockIdx.x * 16;
    int t  = threadIdx.x;
    for (int i = t; i < 16 * NF_CH; i += 256)
        s_a[i] = (__bf16)agg[(size_t)r0 * NF_CH + i];
    __syncthreads();

    int wave = t >> 5, lane = t & 31;
    int n16 = lane & 15, mh = (lane >> 4) * 8;
    {
        int nt = wave;                        // C/16 = 8 tiles, one per wave
        v8f acc = {};
#pragma unroll
        for (int kt = 0; kt < NF_CH / 32; ++kt) {
            v16bf a = load_a(s_a + kt * 32, NF_CH);
            v16bf b = load_b(w2 + (size_t)kt * 32 * C_CH + nt * 16, C_CH);
            acc = wmma_bf(a, b, acc);
        }
        int n = nt * 16 + n16;
        float bias = b2[n];
#pragma unroll
        for (int v = 0; v < 8; ++v)
            s_t[(v + mh) * C_CH + n] = (__bf16)sspf(acc[v] + bias);
    }
    __syncthreads();
    {
        int nt = wave;
        v8f acc = {};
#pragma unroll
        for (int kt = 0; kt < C_CH / 32; ++kt) {
            v16bf a = load_a(s_t + kt * 32, C_CH);
            v16bf b = load_b(w3 + (size_t)kt * 32 * C_CH + nt * 16, C_CH);
            acc = wmma_bf(a, b, acc);
        }
        int n = nt * 16 + n16;
        float bias = b3[n];
#pragma unroll
        for (int v = 0; v < 8; ++v) {
            float y = acc[v] + bias;
            y = (y > 0.0f) ? y : 0.0f;                  // relu
            size_t idx = (size_t)(r0 + v + mh) * C_CH + n;
            xout[idx] = y + x[idx];                     // residual
        }
    }
}

// ---- e_out = tanh(concat(edge_attr, x_out[row]+x_out[col]) @ W_e + b_e) ---
__global__ void __launch_bounds__(256)
k_edge_mlp(const float* __restrict__ edge_attr,
           const int*   __restrict__ eidx,
           const float* __restrict__ xout,
           const __bf16* __restrict__ we, const float* __restrict__ be,
           float* __restrict__ eout, int E) {
    __shared__ __align__(16) __bf16 s_e[16 * EIN_CH]; // 6 KB
    int e0 = blockIdx.x * 16;
    int t  = threadIdx.x;
    // first 64 cols: edge_attr
    for (int i = t; i < 16 * EC_CH; i += 256) {
        int m = i >> 6, c = i & 63;
        s_e[m * EIN_CH + c] = (__bf16)edge_attr[(size_t)e0 * EC_CH + i];
    }
    // last 128 cols: x_out[row] + x_out[col]
    {
        int m = t >> 4;
        int c0 = (t & 15) * 8;
        int r = eidx[e0 + m], c = eidx[E + e0 + m];
        const float* pr = xout + (size_t)r * C_CH + c0;
        const float* pc = xout + (size_t)c * C_CH + c0;
#pragma unroll
        for (int j = 0; j < 8; ++j)
            s_e[m * EIN_CH + EC_CH + c0 + j] = (__bf16)(pr[j] + pc[j]);
    }
    __syncthreads();

    int wave = t >> 5, lane = t & 31;
    int n16 = lane & 15, mh = (lane >> 4) * 8;
    int nt = wave;                                  // 8 N-tiles = 128 cols
    v8f acc = {};
#pragma unroll
    for (int kt = 0; kt < EIN_CH / 32; ++kt) {      // 6 WMMAs
        v16bf a = load_a(s_e + kt * 32, EIN_CH);
        v16bf b = load_b(we + (size_t)kt * 32 * C_CH + nt * 16, C_CH);
        acc = wmma_bf(a, b, acc);
    }
    int n = nt * 16 + n16;
    float bias = be[n];
#pragma unroll
    for (int v = 0; v < 8; ++v)
        eout[(size_t)(e0 + v + mh) * C_CH + n] = tanhf(acc[v] + bias);
}

// ---------------------------------------------------------------------------
extern "C" void kernel_launch(void* const* d_in, const int* in_sizes, int n_in,
                              void* d_out, int out_size, void* d_ws, size_t ws_size,
                              hipStream_t stream) {
    (void)n_in; (void)out_size; (void)ws_size;
    const float* x         = (const float*)d_in[0];
    const int*   eidx      = (const int*)d_in[1];
    const float* edge_attr = (const float*)d_in[2];
    /* d_in[3] = x_pos (unused by reference math) */
    const float* W_f1   = (const float*)d_in[4];
    const float* b_f1   = (const float*)d_in[5];
    const float* W_f2   = (const float*)d_in[6];
    const float* b_f2   = (const float*)d_in[7];
    const float* W_lin1 = (const float*)d_in[8];
    const float* W_lin2 = (const float*)d_in[9];
    const float* b_lin2 = (const float*)d_in[10];
    const float* W_lin3 = (const float*)d_in[11];
    const float* b_lin3 = (const float*)d_in[12];
    const float* W_e    = (const float*)d_in[13];
    const float* b_e    = (const float*)d_in[14];

    const int N = in_sizes[0] / C_CH;   // 50000 (multiple of 16)
    const int E = in_sizes[1] / 2;      // 600000 (multiple of 16)

    float* xout = (float*)d_out;
    float* eout = (float*)d_out + (size_t)N * C_CH;

    // workspace carve-up (256B aligned)
    char* wsp = (char*)d_ws;
    auto carve = [&](size_t bytes) {
        char* p = wsp; wsp += (bytes + 255) & ~(size_t)255; return p;
    };
    __bf16* x_bf = (__bf16*)carve((size_t)N * C_CH  * 2);  // 12.8 MB
    __bf16* h_bf = (__bf16*)carve((size_t)N * NF_CH * 2);  // 25.6 MB
    float*  agg  = (float*) carve((size_t)N * NF_CH * 4);  // 51.2 MB
    __bf16* wf1b = (__bf16*)carve((size_t)EC_CH  * NF_CH * 2);
    __bf16* wf2b = (__bf16*)carve((size_t)NF_CH  * NF_CH * 2);
    __bf16* wl1b = (__bf16*)carve((size_t)C_CH   * NF_CH * 2);
    __bf16* wl2b = (__bf16*)carve((size_t)NF_CH  * C_CH  * 2);
    __bf16* wl3b = (__bf16*)carve((size_t)C_CH   * C_CH  * 2);
    __bf16* web  = (__bf16*)carve((size_t)EIN_CH * C_CH  * 2);

    auto cvt = [&](const float* s, __bf16* d, int n) {
        k_f32_to_bf16<<<(n / 4 + 255) / 256, 256, 0, stream>>>(s, d, n);
    };
    cvt(x,      x_bf, N * C_CH);
    cvt(W_f1,   wf1b, EC_CH  * NF_CH);
    cvt(W_f2,   wf2b, NF_CH  * NF_CH);
    cvt(W_lin1, wl1b, C_CH   * NF_CH);
    cvt(W_lin2, wl2b, NF_CH  * C_CH);
    cvt(W_lin3, wl3b, C_CH   * C_CH);
    cvt(W_e,    web,  EIN_CH * C_CH);

    int aggN = N * NF_CH;
    k_zero_f32<<<(aggN + 255) / 256, 256, 0, stream>>>(agg, aggN);

    k_lin1<<<N / 16, 128, 0, stream>>>(x_bf, wl1b, h_bf);
    k_edge_cfconv<<<E / 16, 256, 0, stream>>>(edge_attr, eidx, h_bf,
                                              wf1b, b_f1, wf2b, b_f2, agg, E);
    k_node_out<<<N / 16, 256, 0, stream>>>(agg, wl2b, b_lin2, wl3b, b_lin3,
                                           x, xout);
    k_edge_mlp<<<E / 16, 256, 0, stream>>>(edge_attr, eidx, xout, web, b_e,
                                           eout, E);
}
